// MambaBlock2D_24670292148451
// MI455X (gfx1250) — matrix operand order repaired and checked by reading, not compiled, not run on hardware
//
#include <hip/hip_runtime.h>
#include <math.h>

// ---------------------------------------------------------------------------
// Types for CDNA5 WMMA (gfx1250, wave32)
// ---------------------------------------------------------------------------
typedef __attribute__((ext_vector_type(16))) __bf16 v16bf;
typedef __attribute__((ext_vector_type(8)))  float  v8f;
typedef unsigned short ushort_t;

// float -> bf16 bits, round-to-nearest-even (off the GEMM hot loop)
static __device__ __forceinline__ ushort_t f2bfu(float f) {
  union { float f; unsigned u; } cv; cv.f = f;
  unsigned u = cv.u + 0x7FFFu + ((cv.u >> 16) & 1u);
  return (ushort_t)(u >> 16);
}

// ---------------------------------------------------------------------------
// Conversion / padding / zero kernels (run once per call; tiny)
// ---------------------------------------------------------------------------
__global__ void cvt_f32_bf16_kernel(const float* __restrict__ in,
                                    ushort_t* __restrict__ out, int n) {
  int i = blockIdx.x * blockDim.x + threadIdx.x;
  if (i < n) out[i] = f2bfu(in[i]);
}

// pad rows: out[(b*rowsOut + r)*cols + c] = r<rowsIn ? bf(in[..]) : 0
__global__ void cvt_pad_rows_kernel(const float* __restrict__ in,
                                    ushort_t* __restrict__ out,
                                    int batch, int rowsIn, int rowsOut, int cols) {
  int idx = blockIdx.x * blockDim.x + threadIdx.x;
  int total = batch * rowsOut * cols;
  if (idx >= total) return;
  int c = idx % cols; int t = idx / cols;
  int r = t % rowsOut; int b = t / rowsOut;
  float v = (r < rowsIn) ? in[((size_t)b * rowsIn + r) * cols + c] : 0.f;
  out[((size_t)b * rowsOut + r) * cols + c] = f2bfu(v);
}

// pad cols: out[r*colsOut + c] = c<colsIn ? bf(in[r*colsIn+c]) : 0
__global__ void cvt_pad_cols_kernel(const float* __restrict__ in,
                                    ushort_t* __restrict__ out,
                                    int rows, int colsIn, int colsOut) {
  int idx = blockIdx.x * blockDim.x + threadIdx.x;
  int total = rows * colsOut;
  if (idx >= total) return;
  int c = idx % colsOut; int r = idx / colsOut;
  float v = (c < colsIn) ? in[(size_t)r * colsIn + c] : 0.f;
  out[(size_t)r * colsOut + c] = f2bfu(v);
}

__global__ void zero_u32_kernel(unsigned* __restrict__ p, int n) {
  int i = blockIdx.x * blockDim.x + threadIdx.x;
  if (i < n) p[i] = 0u;
}

// ---------------------------------------------------------------------------
// Depthwise 3x3 conv (SAME, groups=C), channel-last; optional silu, residual,
// bf16 shadow output.
// ---------------------------------------------------------------------------
__global__ void dwconv_kernel(const float* __restrict__ in, int ldin,
                              const float* __restrict__ w,
                              const float* __restrict__ bias,
                              const float* __restrict__ resid, int ldres,
                              float* __restrict__ out, int ldout,
                              ushort_t* __restrict__ out_bf,
                              int Bn, int H, int W, int Cc, int act) {
  int idx = blockIdx.x * blockDim.x + threadIdx.x;
  int total = Bn * H * W * Cc;
  if (idx >= total) return;
  int c = idx % Cc; int t = idx / Cc;
  int x = t % W; t /= W;
  int y = t % H; int b = t / H;
  float acc = bias ? bias[c] : 0.f;
  const float* wc = w + c * 9;
  #pragma unroll
  for (int ky = 0; ky < 3; ky++) {
    int yy = y + ky - 1;
    if (yy < 0 || yy >= H) continue;
    #pragma unroll
    for (int kx = 0; kx < 3; kx++) {
      int xx = x + kx - 1;
      if (xx < 0 || xx >= W) continue;
      acc += in[((size_t)(b * H + yy) * W + xx) * ldin + c] * wc[ky * 3 + kx];
    }
  }
  if (act == 1) acc = acc / (1.f + __expf(-acc));
  size_t pix = (size_t)(b * H + y) * W + x;
  if (resid) acc += resid[pix * ldres + c];
  if (out)    out[pix * ldout + c] = acc;
  if (out_bf) out_bf[pix * ldout + c] = f2bfu(acc);
}

// ---------------------------------------------------------------------------
// LayerNorm over last dim, one wave32 per row. fp32 and/or bf16 output.
// ---------------------------------------------------------------------------
__global__ void ln_kernel(const float* __restrict__ in,
                          const float* __restrict__ g,
                          const float* __restrict__ b,
                          float* __restrict__ out,
                          ushort_t* __restrict__ out_bf,
                          int rows, int C, float eps) {
  int wid = (blockIdx.x * blockDim.x + threadIdx.x) >> 5;
  int lane = threadIdx.x & 31;
  if (wid >= rows) return;
  const float* r = in + (size_t)wid * C;
  float s = 0.f;
  for (int c = lane; c < C; c += 32) s += r[c];
  #pragma unroll
  for (int o = 16; o > 0; o >>= 1) s += __shfl_xor(s, o, 32);
  float mean = s / (float)C;
  float v = 0.f;
  for (int c = lane; c < C; c += 32) { float d = r[c] - mean; v += d * d; }
  #pragma unroll
  for (int o = 16; o > 0; o >>= 1) v += __shfl_xor(v, o, 32);
  float rstd = rsqrtf(v / (float)C + eps);
  for (int c = lane; c < C; c += 32) {
    float val = (r[c] - mean) * rstd * g[c] + b[c];
    if (out)    out[(size_t)wid * C + c] = val;
    if (out_bf) out_bf[(size_t)wid * C + c] = f2bfu(val);
  }
}

// ---------------------------------------------------------------------------
// bf16 WMMA GEMM, branch-free inner loop.
//   C[m,n] = epi( sum_k A[m,k] * W[n,k] )
// Requirements (guaranteed by caller):
//   M % 64 == 0, Kd % 32 == 0, W buffer padded to ceil(N/16)*16 rows.
// Each wave computes a 64x16 tile (4 M-subtiles sharing one B fragment per
// K-step -> 4 independent WMMAs per iteration).
// epi: v = acc + bias[n]; v = act(v); v += resid; fp32 store (n<N) and/or
// bf16 store (n<bfNcap, stride ldcbf).
// act: 0=none, 1=silu, 2=softplus, 3=gelu(exact)
// Layouts: A 16x32: elem j -> K = k0 + (hi?8:0) + (j<8? j : j+8)
//          B 32x16: elem j -> K = k0 + (hi?8:0) + (j<8? j : j+8)
//          C/D f32: vgpr r -> M = r + (hi?8:0), N = lane&15
// ---------------------------------------------------------------------------
__global__ void gemm_bf16_wmma(const ushort_t* __restrict__ A, int lda,
                               const ushort_t* __restrict__ Wt, int ldw,
                               float* __restrict__ C, int ldc,
                               ushort_t* __restrict__ Cbf, int ldcbf, int bfNcap,
                               const float* __restrict__ bias,
                               const float* __restrict__ resid, int ldres,
                               int M, int N, int Kd, int act,
                               int tilesN, int totalTiles) {
  int wid = blockIdx.x * (blockDim.x >> 5) + (threadIdx.x >> 5);
  if (wid >= totalTiles) return;              // whole-wave uniform exit
  int lane = threadIdx.x & 31;
  int lrow = lane & 15;
  bool hi = lane >= 16;
  int m0 = (wid / tilesN) * 64;
  int n0 = (wid % tilesN) * 16;
  const ushort_t* aptr = A  + (size_t)(m0 + lrow) * lda + (hi ? 8 : 0);
  const ushort_t* wptr = Wt + (size_t)(n0 + lrow) * ldw + (hi ? 8 : 0);
  size_t arow = (size_t)lda * 16;             // elements between M-subtiles
  union FragU { v16bf v; uint4 q[2]; };
  v8f acc0 = {}, acc1 = {}, acc2 = {}, acc3 = {};
  for (int k0 = 0; k0 < Kd; k0 += 32) {
    FragU fb, fa0, fa1, fa2, fa3;
    fb.q[0]  = *(const uint4*)(wptr + k0);
    fb.q[1]  = *(const uint4*)(wptr + k0 + 16);
    fa0.q[0] = *(const uint4*)(aptr + k0);
    fa0.q[1] = *(const uint4*)(aptr + k0 + 16);
    fa1.q[0] = *(const uint4*)(aptr + arow + k0);
    fa1.q[1] = *(const uint4*)(aptr + arow + k0 + 16);
    fa2.q[0] = *(const uint4*)(aptr + 2 * arow + k0);
    fa2.q[1] = *(const uint4*)(aptr + 2 * arow + k0 + 16);
    fa3.q[0] = *(const uint4*)(aptr + 3 * arow + k0);
    fa3.q[1] = *(const uint4*)(aptr + 3 * arow + k0 + 16);
    acc0 = __builtin_amdgcn_wmma_f32_16x16x32_bf16(false, fa0.v, false, fb.v,
                                                   (short)0, acc0, false, false);
    acc1 = __builtin_amdgcn_wmma_f32_16x16x32_bf16(false, fa1.v, false, fb.v,
                                                   (short)0, acc1, false, false);
    acc2 = __builtin_amdgcn_wmma_f32_16x16x32_bf16(false, fa2.v, false, fb.v,
                                                   (short)0, acc2, false, false);
    acc3 = __builtin_amdgcn_wmma_f32_16x16x32_bf16(false, fa3.v, false, fb.v,
                                                   (short)0, acc3, false, false);
  }
  int nn = n0 + lrow;
  #pragma unroll
  for (int i = 0; i < 4; i++) {
    v8f acc = (i == 0) ? acc0 : (i == 1) ? acc1 : (i == 2) ? acc2 : acc3;
    #pragma unroll
    for (int r = 0; r < 8; r++) {
      int mm = m0 + i * 16 + r + (hi ? 8 : 0);
      float v = acc[r];
      if (bias && nn < N) v += bias[nn];
      if (act == 1) v = v / (1.f + __expf(-v));
      else if (act == 2) v = (v > 20.f) ? v : log1pf(__expf(v));
      else if (act == 3) v = 0.5f * v * (1.f + erff(v * 0.70710678118654752f));
      if (resid && nn < N) v += resid[(size_t)mm * ldres + nn];
      if (C && nn < N)      C[(size_t)mm * ldc + nn] = v;
      if (Cbf && nn < bfNcap) Cbf[(size_t)mm * ldcbf + nn] = f2bfu(v);
    }
  }
}

// ---------------------------------------------------------------------------
// (h,w)->(w,h) reorder, fp32 + bf16 outputs
// ---------------------------------------------------------------------------
__global__ void transpose_hw_kernel(const float* __restrict__ in,
                                    float* __restrict__ out,
                                    ushort_t* __restrict__ out_bf,
                                    int Bn, int H, int W, int Cc) {
  int idx = blockIdx.x * blockDim.x + threadIdx.x;
  int total = Bn * H * W * Cc;
  if (idx >= total) return;
  int c = idx % Cc; int t = idx / Cc;
  int w = t % W; t /= W;
  int h = t % H; int b = t / H;
  size_t L = (size_t)H * W;
  float v = in[((size_t)b * L + (size_t)h * W + w) * Cc + c];
  size_t o = ((size_t)b * L + (size_t)w * H + h) * Cc + c;
  out[o] = v;
  out_bf[o] = f2bfu(v);
}

// ---------------------------------------------------------------------------
// Selective scan. One block per (k,b), one thread per channel d.
// Reversed directions (k>=2) walk backward and store at the un-reversed
// position, so combine needs no flipping.
// ---------------------------------------------------------------------------
__global__ void scan_kernel(const float* __restrict__ xxc,   // (B,L,Di) row order
                            const float* __restrict__ xsB,   // (B,L,Di) col order
                            const float* __restrict__ dbl,   // (K,B,L,44)
                            const float* __restrict__ dtb,   // (K,B,L,Di)
                            const float* __restrict__ A_logs,// (K*Di,16)
                            const float* __restrict__ Ds,    // (K*Di)
                            float* __restrict__ ys,          // (K,B,L,Di)
                            int Bn, int L, int Di, int Rr, int dblld) {
  int k = blockIdx.x / Bn;
  int b = blockIdx.x % Bn;
  int d = threadIdx.x;
  bool rev = (k >= 2);
  const float* u  = ((k & 1) ? xsB : xxc) + (size_t)b * L * Di + d;
  const float* dt = dtb + ((size_t)(k * Bn + b) * L) * Di + d;
  const float* bc = dbl + ((size_t)(k * Bn + b) * L) * dblld;
  float* yo       = ys  + ((size_t)(k * Bn + b) * L) * Di + d;
  float Av[16];
  #pragma unroll
  for (int n = 0; n < 16; n++) Av[n] = -__expf(A_logs[(size_t)(k * Di + d) * 16 + n]);
  float Dv = Ds[k * Di + d];
  float h[16];
  #pragma unroll
  for (int n = 0; n < 16; n++) h[n] = 0.f;
  int p = rev ? (L - 1) : 0;
  int stepdir = rev ? -1 : 1;
  for (int t = 0; t < L; t++, p += stepdir) {
    float uv  = u[(size_t)p * Di];
    float dtv = dt[(size_t)p * Di];
    const float4* bp = (const float4*)(bc + (size_t)p * dblld + Rr); // 16B aligned
    float Bv[16], Cv[16];
    ((float4*)Bv)[0] = bp[0]; ((float4*)Bv)[1] = bp[1];
    ((float4*)Bv)[2] = bp[2]; ((float4*)Bv)[3] = bp[3];
    ((float4*)Cv)[0] = bp[4]; ((float4*)Cv)[1] = bp[5];
    ((float4*)Cv)[2] = bp[6]; ((float4*)Cv)[3] = bp[7];
    float du = dtv * uv;
    float y = uv * Dv;
    #pragma unroll
    for (int n = 0; n < 16; n++) {
      float dA = __expf(dtv * Av[n]);
      h[n] = dA * h[n] + du * Bv[n];
      y += h[n] * Cv[n];
    }
    yo[(size_t)p * Di] = y;
  }
}

// ---------------------------------------------------------------------------
// Combine 4 directions back into row order: out[b, h*W+w, d]
// ---------------------------------------------------------------------------
__global__ void combine_kernel(const float* __restrict__ ys,
                               float* __restrict__ out,
                               int Bn, int H, int W, int Di) {
  int idx = blockIdx.x * blockDim.x + threadIdx.x;
  int total = Bn * H * W * Di;
  if (idx >= total) return;
  int d = idx % Di; int t = idx / Di;
  int w = t % W; t /= W;
  int h = t % H; int b = t / H;
  size_t L = (size_t)H * W;
  size_t la = (size_t)h * W + w;
  size_t lb = (size_t)w * H + h;
  size_t s0 = ((size_t)(0 * Bn + b) * L + la) * Di + d;
  size_t s1 = ((size_t)(1 * Bn + b) * L + lb) * Di + d;
  size_t s2 = ((size_t)(2 * Bn + b) * L + la) * Di + d;
  size_t s3 = ((size_t)(3 * Bn + b) * L + lb) * Di + d;
  out[((size_t)b * L + la) * Di + d] = ys[s0] + ys[s1] + ys[s2] + ys[s3];
}

// ---------------------------------------------------------------------------
// Gate: out_bf[m,d] = yln[m,d] * silu(z[m,d]),  z = xz[:, Di:2Di]
// ---------------------------------------------------------------------------
__global__ void gate_kernel(const float* __restrict__ yln,
                            const float* __restrict__ xz,
                            ushort_t* __restrict__ out_bf,
                            int Mrows, int Di, int ldz) {
  int idx = blockIdx.x * blockDim.x + threadIdx.x;
  int total = Mrows * Di;
  if (idx >= total) return;
  int d = idx % Di; int m = idx / Di;
  float z = xz[(size_t)m * ldz + Di + d];
  out_bf[idx] = f2bfu(yln[idx] * (z / (1.f + __expf(-z))));
}

// ---------------------------------------------------------------------------
// Host-side launcher
// ---------------------------------------------------------------------------
static inline void launch_gemm(const ushort_t* A, int lda, const ushort_t* Wt, int ldw,
                               float* C, int ldc, ushort_t* Cbf, int ldcbf, int bfNcap,
                               const float* bias, const float* resid, int ldres,
                               int M, int N, int K, int act, hipStream_t s) {
  int tm = M / 64;                    // M % 64 == 0 guaranteed
  int tn = (N + 15) / 16;
  int tiles = tm * tn;
  int blocks = (tiles + 3) / 4;       // 4 waves per 128-thread block
  gemm_bf16_wmma<<<dim3(blocks), dim3(128), 0, s>>>(
      A, lda, Wt, ldw, C, ldc, Cbf, ldcbf, bfNcap, bias, resid, ldres,
      M, N, K, act, tn, tiles);
}

extern "C" void kernel_launch(void* const* d_in, const int* in_sizes, int n_in,
                              void* d_out, int out_size, void* d_ws, size_t ws_size,
                              hipStream_t stream) {
  // ---- problem dims ----
  const int Bn = 2, H = 56, W = 56, Cch = 192;
  const int Di = 384, Kdir = 4, Rr = 12;
  const int hid = 768, two = 768;           // 2*Di
  const int L = H * W;                      // 3136
  const int BL = Bn * L;                    // 6272 (div by 64)
  const int dblld = 44;                     // R + 2N = 12 + 32
  const int Npad = 48;                      // x_proj weight rows padded
  const int Kpad = 32;                      // dt GEMM K padded

  // ---- inputs (setup_inputs order) ----
  const float* x        = (const float*)d_in[0];
  const float* cpe1_w   = (const float*)d_in[1];
  const float* cpe1_b   = (const float*)d_in[2];
  const float* ln1_g    = (const float*)d_in[3];
  const float* ln1_b    = (const float*)d_in[4];
  const float* in_proj_w= (const float*)d_in[5];
  const float* conv_w   = (const float*)d_in[6];
  const float* conv_b   = (const float*)d_in[7];
  const float* x_proj_w = (const float*)d_in[8];
  const float* dt_w     = (const float*)d_in[9];
  const float* dt_b     = (const float*)d_in[10];
  const float* A_logs   = (const float*)d_in[11];
  const float* Ds       = (const float*)d_in[12];
  const float* onorm_g  = (const float*)d_in[13];
  const float* onorm_b  = (const float*)d_in[14];
  const float* out_proj_w = (const float*)d_in[15];
  const float* cpe2_w   = (const float*)d_in[16];
  const float* cpe2_b   = (const float*)d_in[17];
  const float* ln2_g    = (const float*)d_in[18];
  const float* ln2_b    = (const float*)d_in[19];
  const float* fc1_w    = (const float*)d_in[20];
  const float* fc1_b    = (const float*)d_in[21];
  const float* fc2_w    = (const float*)d_in[22];
  const float* fc2_b    = (const float*)d_in[23];

  // ---- workspace layout (32-byte aligned blocks) ----
  float* ws = (float*)d_ws;
  size_t off = 0;
  auto afl = [&](size_t nfl) { float* p = ws + off; off += (nfl + 7) & ~(size_t)7; return p; };
  auto ash = [&](size_t nsh) { return (ushort_t*)afl((nsh + 1) / 2); };

  // fp32 buffers
  float* t0   = afl((size_t)BL * Cch);        // x + cpe1
  float* xz   = afl((size_t)BL * two);        // in_proj out (xx|z)
  float* xxc  = afl((size_t)BL * Di);         // conv+silu out; later ycomb
  float* xsB  = afl((size_t)BL * Di);         // transposed xx; later yln
  float* dbl  = afl((size_t)Kdir * BL * dblld);
  float* dtb  = afl((size_t)Kdir * BL * Di);  // later holds h1_bf (as ushort)
  float* ys   = afl((size_t)Kdir * BL * Di);  // later x2 / t3
  // bf16 buffers
  ushort_t* t1_bf   = ash((size_t)BL * Cch);  // LN1 out; later LN2 out
  ushort_t* xxc_bf  = ash((size_t)BL * Di);   // conv out bf16; later gate out
  ushort_t* xsB_bf  = ash((size_t)BL * Di);
  ushort_t* dtin_bf = ash((size_t)Kdir * BL * Kpad); // padded dt-GEMM input
  // bf16 weights (padded where needed)
  ushort_t* w_in_bf  = ash((size_t)two * Cch);
  ushort_t* w_xpp_bf = ash((size_t)Kdir * Npad * Di);   // rows 44->48 padded
  ushort_t* w_dtp_bf = ash((size_t)Kdir * Di * Kpad);   // cols 12->32 padded
  ushort_t* w_op_bf  = ash((size_t)Cch * Di);
  ushort_t* w_f1_bf  = ash((size_t)hid * Cch);
  ushort_t* w_f2_bf  = ash((size_t)Cch * hid);

  float*    ycomb   = xxc;
  float*    yln     = xsB;
  ushort_t* ln2_bf  = t1_bf;
  ushort_t* gate_bf = xxc_bf;
  ushort_t* h1_bf   = (ushort_t*)dtb;         // dtb dead after scan
  float*    x2      = ys;                      // ys dead after combine
  float*    t3      = ys + (size_t)BL * Cch;
  float*    dout    = (float*)d_out;

  const int TPB = 256;
  #define GRID(n) dim3((unsigned)(((n) + TPB - 1) / TPB))

  // 0) weight conversions (plain + padded) and dtin zero-fill
  cvt_f32_bf16_kernel<<<GRID(two * Cch), TPB, 0, stream>>>(in_proj_w, w_in_bf, two * Cch);
  cvt_pad_rows_kernel<<<GRID(Kdir * Npad * Di), TPB, 0, stream>>>(
      x_proj_w, w_xpp_bf, Kdir, dblld, Npad, Di);
  cvt_pad_cols_kernel<<<GRID(Kdir * Di * Kpad), TPB, 0, stream>>>(
      dt_w, w_dtp_bf, Kdir * Di, Rr, Kpad);
  cvt_f32_bf16_kernel<<<GRID(Cch * Di), TPB, 0, stream>>>(out_proj_w, w_op_bf, Cch * Di);
  cvt_f32_bf16_kernel<<<GRID(hid * Cch), TPB, 0, stream>>>(fc1_w, w_f1_bf, hid * Cch);
  cvt_f32_bf16_kernel<<<GRID(Cch * hid), TPB, 0, stream>>>(fc2_w, w_f2_bf, Cch * hid);
  {
    int nz = (Kdir * BL * Kpad) / 2;  // ushorts -> u32 words
    zero_u32_kernel<<<GRID(nz), TPB, 0, stream>>>((unsigned*)dtin_bf, nz);
  }

  // 1) t0 = x + cpe1(x)
  dwconv_kernel<<<GRID(BL * Cch), TPB, 0, stream>>>(
      x, Cch, cpe1_w, cpe1_b, x, Cch, t0, Cch, nullptr, Bn, H, W, Cch, 0);

  // 2) t1_bf = LN(t0) (bf16 only; consumed solely by in_proj GEMM)
  ln_kernel<<<GRID(BL * 32), 128, 0, stream>>>(t0, ln1_g, ln1_b, nullptr, t1_bf,
                                               BL, Cch, 1e-6f);

  // 3) xz = t1 @ in_proj_w^T   (M=BL, N=768, K=192)
  launch_gemm(t1_bf, Cch, w_in_bf, Cch, xz, two, nullptr, 0, 0,
              nullptr, nullptr, 0, BL, two, Cch, 0, stream);

  // 4) xxc(+bf) = silu(dwconv(xx)), xx = xz[:, :Di] viewed with ldin=768
  dwconv_kernel<<<GRID(BL * Di), TPB, 0, stream>>>(
      xz, two, conv_w, conv_b, nullptr, 0, xxc, Di, xxc_bf, Bn, H, W, Di, 1);

  // 5) xsB(+bf) = (h,w)->(w,h) reorder of xxc
  transpose_hw_kernel<<<GRID(BL * Di), TPB, 0, stream>>>(xxc, xsB, xsB_bf, Bn, H, W, Di);

  // 6) per-direction x_proj (N=44, K=384; padded weights):
  //    fp32 -> dbl (ld 44); bf16 of first 12 cols -> dtin_bf (ld 32)
  for (int k = 0; k < Kdir; k++) {
    const ushort_t* Aord = (k & 1) ? xsB_bf : xxc_bf;
    launch_gemm(Aord, Di, w_xpp_bf + (size_t)k * Npad * Di, Di,
                dbl + (size_t)k * BL * dblld, dblld,
                dtin_bf + (size_t)k * BL * Kpad, Kpad, Rr,
                nullptr, nullptr, 0, BL, dblld, Di, 0, stream);
  }

  // 7) per-direction dt: dtb_k = softplus(dtin_k @ dt_w[k]^T + dt_b[k])
  //    (K padded to 32 -> single branch-free K-step)
  for (int k = 0; k < Kdir; k++) {
    launch_gemm(dtin_bf + (size_t)k * BL * Kpad, Kpad,
                w_dtp_bf + (size_t)k * Di * Kpad, Kpad,
                dtb + (size_t)k * BL * Di, Di, nullptr, 0, 0,
                dt_b + (size_t)k * Di, nullptr, 0,
                BL, Di, Kpad, 2, stream);
  }

  // 8) selective scan: one block per (k,b), Di threads
  scan_kernel<<<dim3(Kdir * Bn), dim3(Di), 0, stream>>>(
      xxc, xsB, dbl, dtb, A_logs, Ds, ys, Bn, L, Di, Rr, dblld);

  // 9) combine 4 directions -> ycomb (B,L,Di)
  combine_kernel<<<GRID(BL * Di), TPB, 0, stream>>>(ys, ycomb, Bn, H, W, Di);

  // 10) out-norm LN over Di (eps 1e-5), fp32 (gate reads it)
  ln_kernel<<<GRID(BL * 32), 128, 0, stream>>>(ycomb, onorm_g, onorm_b, yln, nullptr,
                                               BL, Di, 1e-5f);

  // 11) gate_bf = yln * silu(z)
  gate_kernel<<<GRID(BL * Di), TPB, 0, stream>>>(yln, xz, gate_bf, BL, Di, two);

  // 12) x2 = t0 + gate @ out_proj_w^T  (M=BL, N=192, K=384)
  launch_gemm(gate_bf, Di, w_op_bf, Di, x2, Cch, nullptr, 0, 0,
              nullptr, t0, Cch, BL, Cch, Di, 0, stream);

  // 13) t3 = x2 + cpe2(x2)
  dwconv_kernel<<<GRID(BL * Cch), TPB, 0, stream>>>(
      x2, Cch, cpe2_w, cpe2_b, x2, Cch, t3, Cch, nullptr, Bn, H, W, Cch, 0);

  // 14) ln2_bf = LN(t3) (bf16 only; consumed solely by fc1)
  ln_kernel<<<GRID(BL * 32), 128, 0, stream>>>(t3, ln2_g, ln2_b, nullptr, ln2_bf,
                                               BL, Cch, 1e-6f);

  // 15) h1_bf = gelu(ln2 @ fc1_w^T + fc1_b)  (N=768, K=192; bf16-only output)
  launch_gemm(ln2_bf, Cch, w_f1_bf, Cch, nullptr, 0, h1_bf, hid, hid,
              fc1_b, nullptr, 0, BL, hid, Cch, 3, stream);

  // 16) dout = t3 + (h1 @ fc2_w^T + fc2_b)  (N=192, K=768)
  launch_gemm(h1_bf, hid, w_f2_bf, hid, dout, Cch, nullptr, 0, 0,
              fc2_b, t3, Cch, BL, Cch, hid, 0, stream);

  #undef GRID
  (void)in_sizes; (void)n_in; (void)out_size; (void)ws_size;
}
